// GRPENetwork_30691836297648
// MI455X (gfx1250) — matrix operand order repaired and checked by reading, not compiled, not run on hardware
//
#include <hip/hip_runtime.h>
#include <hip/hip_bf16.h>
#include <math.h>

// ---------------------------------------------------------------------------
// CDNA5 (gfx1250) GRPE encoder layer: all GEMMs via v_wmma_f32_16x16x32_bf16.
// wave32; 8 waves / 256-thread block; each wave computes a (16*TM)x(16*TN) C
// tile (register blocking both ways: A frags reused across TN, B frags across
// TM).  All B operands are pre-laid-out K-contiguous so fragment loads are
// b128.
// ---------------------------------------------------------------------------

typedef __bf16 v16bf __attribute__((ext_vector_type(16)));
typedef float  v8f   __attribute__((ext_vector_type(8)));

#define DEVI static __device__ __forceinline__

DEVI unsigned short f2bf(float f) {
  union { float f; unsigned u; } v; v.f = f;
  unsigned u = v.u;
  return (unsigned short)((u + 0x7fffu + ((u >> 16) & 1u)) >> 16); // RNE
}
DEVI float bf2f(unsigned short h) {
  union { unsigned u; float f; } v; v.u = ((unsigned)h) << 16;
  return v.f;
}
DEVI float gelu_exact(float x) {
  return 0.5f * x * (1.0f + erff(x * 0.70710678118654752440f));
}

// ---------------------------- prep / relayout kernels -----------------------
__global__ void cvt_bf16_k(const float* __restrict__ s,
                           unsigned short* __restrict__ d, int n) {
  int i = blockIdx.x * 256 + threadIdx.x;
  if (i < n) d[i] = f2bf(s[i]);
}

// weight transpose: in f32 [K,N] row-major -> out bf16 [N,K] (K-contiguous)
__global__ void wt_t_k(const float* __restrict__ in, unsigned short* __restrict__ out,
                       int K, int N) {
  int i = blockIdx.x * 256 + threadIdx.x;
  if (i >= K * N) return;
  int n = i / K, k = i % K;
  out[i] = f2bf(in[(long)k * N + n]);
}

// hop/edge V-table transpose+pad: in f32 [Mbins, 256] (cols = h*32+d)
// -> out bf16 [8][32][Mpad], zero past Mbins (K-contiguous in m).
__global__ void vtab_t_k(const float* __restrict__ in, unsigned short* __restrict__ out,
                         int Mbins, int Mpad) {
  int i = blockIdx.x * 256 + threadIdx.x;
  if (i >= 8 * 32 * Mpad) return;
  int h = i / (32 * Mpad);
  int rem = i % (32 * Mpad);
  int d = rem / Mpad, m = rem % Mpad;
  out[i] = (m < Mbins) ? f2bf(in[(long)m * 256 + h * 32 + d]) : (unsigned short)0;
}

// v transpose: in bf16 [b,n,h,d] ([8192,256]) -> out bf16 [bh][d][n] ([128][32][512])
__global__ void v_t_k(const unsigned short* __restrict__ in,
                      unsigned short* __restrict__ out) {
  int i = blockIdx.x * 256 + threadIdx.x;        // over 128*32*512 = 2097152
  int bh = i >> 14;                              // /16384
  int rem = i & 16383;
  int d = rem >> 9, n = rem & 511;
  int b = bh >> 3, h = bh & 7;
  out[i] = in[(((long)b * 512 + n) * 256) + h * 32 + d];
}

// ------------------------------ generic WMMA GEMM ---------------------------
// C[M,N] = A[M,K](bf16 row-major) x B (bf16, K-contiguous: element (k,n) at
// B[n*ldb + k]) (+bias)(+residual)(gelu?) -> f32 and/or bf16 outputs.
// Batched: batch = blockIdx.y; bo = batch/inner, bi = batch%inner (element
// strides).  GUARD: branch-free clamped loads + masked stores for padded N.
// Requirements: K multiple of 32; M multiple of 16*TM; lda/ldb multiples of 8
// elements (b128 alignment); for GUARD=false, N multiple of 16*TN.
template <int TM, int TN, bool GUARD>
__global__ void gemm_wmma_k(
    const unsigned short* __restrict__ A, long sAo, long sAi, int lda,
    const unsigned short* __restrict__ B, long sBo, long sBi, int ldb,
    const float* __restrict__ bias,
    const float* __restrict__ resid, long sRo, long sRi, int ldr,
    float* __restrict__ outF, long sFo, long sFi, int ldf,
    unsigned short* __restrict__ outH, long sHo, long sHi, int ldh,
    int M, int Nreal, int K, int inner, int act)
{
  const int batch = blockIdx.y;
  const int bo = batch / inner, bi = batch % inner;
  const unsigned short* Ab = A + bo * sAo + bi * sAi;
  const unsigned short* Bb = B + bo * sBo + bi * sBi;

  const int lane = threadIdx.x & 31;           // wave32 lane id
  const int half = lane >> 4;
  const int l16  = lane & 15;

  const int tiles_n = (Nreal + 15) >> 4;
  const int ngroups = (tiles_n + TN - 1) / TN;
  const int mgroups = M / (16 * TM);
  const int tiles   = mgroups * ngroups;
  const int t = blockIdx.x * 8 + (threadIdx.x >> 5);
  if (t >= tiles) return;
  const int tm = t / ngroups, tg = t % ngroups;

  const unsigned short* Arow[TM];
  #pragma unroll
  for (int i = 0; i < TM; ++i)
    Arow[i] = Ab + (long)(tm * 16 * TM + i * 16 + l16) * lda + 8 * half;

  int  colv[TN];   // real output column for this lane
  bool cbv[TN];
  const unsigned short* Bcol[TN];
  #pragma unroll
  for (int j = 0; j < TN; ++j) {
    colv[j] = (tg * TN + j) * 16 + l16;
    cbv[j]  = GUARD ? (colv[j] < Nreal) : true;
    int cc  = GUARD ? (cbv[j] ? colv[j] : 0) : colv[j];
    Bcol[j] = Bb + (long)cc * ldb + 8 * half;
  }

  union F16 { unsigned u[8]; v16bf v; };
  union F32 { float f[8]; v8f v; };
  F16 a[TM];
  F32 c[TM][TN];
  #pragma unroll
  for (int i = 0; i < TM; ++i)
    #pragma unroll
    for (int j = 0; j < TN; ++j)
      #pragma unroll
      for (int r = 0; r < 8; ++r) c[i][j].f[r] = 0.0f;

  for (int k0 = 0; k0 < K; k0 += 32) {
    if (k0 + 32 < K) __builtin_prefetch(Arow[0] + k0 + 32);  // global_prefetch_b8
    // A frags (16-bit A 16x32): two contiguous 8-element runs -> 2x b128 each
    #pragma unroll
    for (int i = 0; i < TM; ++i) {
      const uint4 a0 = *(const uint4*)(Arow[i] + k0);
      const uint4 a1 = *(const uint4*)(Arow[i] + k0 + 16);
      a[i].u[0] = a0.x; a[i].u[1] = a0.y; a[i].u[2] = a0.z; a[i].u[3] = a0.w;
      a[i].u[4] = a1.x; a[i].u[5] = a1.y; a[i].u[6] = a1.z; a[i].u[7] = a1.w;
    }
    #pragma unroll
    for (int j = 0; j < TN; ++j) {
      // B frag (16-bit B 32x16): two contiguous 8-element runs -> 2x b128
      F16 b;
      const uint4 b0 = *(const uint4*)(Bcol[j] + k0);
      const uint4 b1 = *(const uint4*)(Bcol[j] + k0 + 16);
      b.u[0] = b0.x; b.u[1] = b0.y; b.u[2] = b0.z; b.u[3] = b0.w;
      b.u[4] = b1.x; b.u[5] = b1.y; b.u[6] = b1.z; b.u[7] = b1.w;
      #pragma unroll
      for (int i = 0; i < TM; ++i)
        c[i][j].v = __builtin_amdgcn_wmma_f32_16x16x32_bf16(
            false, a[i].v, false, b.v, (short)0, c[i][j].v, false, false);
    }
  }

  // epilogue
  #pragma unroll
  for (int j = 0; j < TN; ++j) {
    if (GUARD && !cbv[j]) continue;
    const int col = colv[j];
    const float bval = bias ? bias[col] : 0.0f;
    #pragma unroll
    for (int i = 0; i < TM; ++i) {
      #pragma unroll
      for (int r = 0; r < 8; ++r) {
        int rowm = tm * 16 * TM + i * 16 + r + 8 * half;  // C: VGPR r -> row r+8*half
        float v = c[i][j].f[r] + bval;
        if (resid) v += resid[bo * sRo + bi * sRi + (long)rowm * ldr + col];
        if (act == 1) v = gelu_exact(v);
        if (outF) outF[bo * sFo + bi * sFi + (long)rowm * ldf + col] = v;
        if (outH) outH[bo * sHo + bi * sHi + (long)rowm * ldh + col] = f2bf(v);
      }
    }
  }
}

template <int TM, int TN, bool GUARD>
static void run_gemm(hipStream_t stream,
    const unsigned short* A, long sAo, long sAi, int lda,
    const unsigned short* B, long sBo, long sBi, int ldb,
    const float* bias,
    const float* resid, long sRo, long sRi, int ldr,
    float* outF, long sFo, long sFi, int ldf,
    unsigned short* outH, long sHo, long sHi, int ldh,
    int M, int Nreal, int K, int inner, int batches, int act)
{
  int tiles_n = (Nreal + 15) >> 4;
  int ngroups = (tiles_n + TN - 1) / TN;
  int tiles = (M / (16 * TM)) * ngroups;
  dim3 grid((tiles + 7) / 8, batches);
  gemm_wmma_k<TM, TN, GUARD><<<grid, 256, 0, stream>>>(
      A, sAo, sAi, lda, B, sBo, sBi, ldb, bias, resid, sRo, sRi, ldr,
      outF, sFo, sFi, ldf, outH, sHo, sHi, ldh, M, Nreal, K, inner, act);
}

// ------------------------------- LayerNorm ----------------------------------
// One wave per 256-wide row; 8 rows per block.  Output bf16.
__global__ void ln_k(const float* __restrict__ x, const float* __restrict__ g,
                     const float* __restrict__ b, unsigned short* __restrict__ o,
                     int rows) {
  int row = blockIdx.x * 8 + (threadIdx.x >> 5);
  int lane = threadIdx.x & 31;
  if (row >= rows) return;
  const float* xr = x + (long)row * 256;
  float v[8], s = 0.0f;
  #pragma unroll
  for (int t = 0; t < 8; ++t) { v[t] = xr[lane + 32 * t]; s += v[t]; }
  #pragma unroll
  for (int m = 16; m >= 1; m >>= 1) s += __shfl_xor(s, m, 32);
  float mean = s * (1.0f / 256.0f);
  float var = 0.0f;
  #pragma unroll
  for (int t = 0; t < 8; ++t) { float d = v[t] - mean; var += d * d; }
  #pragma unroll
  for (int m = 16; m >= 1; m >>= 1) var += __shfl_xor(var, m, 32);
  float rstd = rsqrtf(var * (1.0f / 256.0f) + 1e-5f);
  #pragma unroll
  for (int t = 0; t < 8; ++t) {
    int j = lane + 32 * t;
    o[(long)row * 256 + j] = f2bf((v[t] - mean) * rstd * g[j] + b[j]);
  }
}

// ------------------- bias gather + mask + scale + softmax -------------------
// One wave per attention row (512 keys, 16 per lane).  Reads raw qk f32 from
// S, adds gathered relative-position biases, scales, masks, softmaxes, and
// writes normalized att as bf16 into the SAME buffer region (bf16 row packed
// into the first half of the row's f32 slot: elem index row*1024 + j).
__global__ void softmax_bias_k(
    float* __restrict__ S, unsigned short* __restrict__ att,
    const unsigned short* __restrict__ qh, const unsigned short* __restrict__ kh,
    const unsigned short* __restrict__ qe, const unsigned short* __restrict__ ke,
    const int* __restrict__ dist, const int* __restrict__ edge,
    const unsigned char* __restrict__ mask)
{
  const int row  = blockIdx.x * 8 + (threadIdx.x >> 5);
  const int lane = threadIdx.x & 31;
  const int bh = row >> 9, i = row & 511, b = bh >> 3;
  const float scale = 0.17677669529663687f;  // 32^-0.5
  const long sbase = (long)row * 512;
  const long dbase = ((long)b * 512 + i) * 512;
  const unsigned short* qhr = qh + ((long)bh * 512 + i) * 258;
  const unsigned short* qer = qe + ((long)bh * 512 + i) * 27;

  float s[16];
  float mx = -INFINITY;
  #pragma unroll
  for (int t = 0; t < 16; ++t) {
    int j = lane + 32 * t;
    int dr = dist[dbase + j]; int d = (dr == -1) ? 257 : (dr > 256 ? 256 : dr);
    int er = edge[dbase + j]; int e = (er == -1) ? 26  : (er > 25  ? 25  : er);
    float bias = bf2f(qhr[d]) + bf2f(kh[((long)bh * 512 + j) * 258 + d])
               + bf2f(qer[e]) + bf2f(ke[((long)bh * 512 + j) * 27  + e]);
    float v = (S[sbase + j] + bias) * scale;
    if (mask[b * 512 + j]) v = -INFINITY;
    s[t] = v;
    mx = fmaxf(mx, v);
  }
  #pragma unroll
  for (int m = 16; m >= 1; m >>= 1) mx = fmaxf(mx, __shfl_xor(mx, m, 32));
  float sum = 0.0f;
  #pragma unroll
  for (int t = 0; t < 16; ++t) { s[t] = __expf(s[t] - mx); sum += s[t]; }
  #pragma unroll
  for (int m = 16; m >= 1; m >>= 1) sum += __shfl_xor(sum, m, 32);
  float inv = 1.0f / sum;
  #pragma unroll
  for (int t = 0; t < 16; ++t)
    att[(long)row * 1024 + lane + 32 * t] = f2bf(s[t] * inv);
}

// ------------------------- scatter att into bins ----------------------------
// One wave per row; per-wave LDS bin array (288 hop bins zero-padded past 258,
// 32 edge bins zero-padded past 27) via ds_add_f32; stores bf16 padded bins so
// the downstream vha@Vh / vea@Ve GEMMs use clean K multiples of 32.
__global__ void scatter_k(const unsigned short* __restrict__ att,
                          const int* __restrict__ dist, const int* __restrict__ edge,
                          unsigned short* __restrict__ vha,
                          unsigned short* __restrict__ vea)
{
  __shared__ float bins[8][320];
  const int w = threadIdx.x >> 5, lane = threadIdx.x & 31;
  const int row = blockIdx.x * 8 + w;
  const int bh = row >> 9, i = row & 511, b = bh >> 3;
  float* bw = bins[w];
  #pragma unroll
  for (int t = 0; t < 10; ++t) bw[lane + 32 * t] = 0.0f;
  __syncthreads();
  const long dbase = ((long)b * 512 + i) * 512;
  const unsigned short* ar = att + (long)row * 1024;
  for (int t = 0; t < 16; ++t) {
    int j = lane + 32 * t;
    float a = bf2f(ar[j]);
    int dr = dist[dbase + j]; int d = (dr == -1) ? 257 : (dr > 256 ? 256 : dr);
    int er = edge[dbase + j]; int e = (er == -1) ? 26  : (er > 25  ? 25  : er);
    atomicAdd(&bw[d], a);
    atomicAdd(&bw[288 + e], a);
  }
  __syncthreads();
  const long ob = (long)row * 288;
  #pragma unroll
  for (int t = 0; t < 9; ++t) { int m = lane + 32 * t; vha[ob + m] = f2bf(bw[m]); }
  vea[(long)row * 32 + lane] = f2bf(bw[288 + lane]);
}

// ---------------------------------------------------------------------------
extern "C" void kernel_launch(void* const* d_in, const int* in_sizes, int n_in,
                              void* d_out, int out_size, void* d_ws, size_t ws_size,
                              hipStream_t stream)
{
  (void)in_sizes; (void)n_in; (void)out_size; (void)ws_size;
  // Problem constants
  const int Bn = 16, Nn = 512, Hn = 8, DM = 256, FFn = 1024, DIN = 128, DOUT = 128;
  const int NHOP = 258, NEDGE = 27, BH = Bn * Hn, ROWS = Bn * Nn; // 128, 8192

  // -------- inputs (setup_inputs order) --------
  const float* x      = (const float*)d_in[0];
  const unsigned char* mask = (const unsigned char*)d_in[1]; // bool mask
  const int*   dist   = (const int*)d_in[2];
  const int*   edge   = (const int*)d_in[3];
  const float* node_W = (const float*)d_in[4];  const float* node_b = (const float*)d_in[5];
  const float* ln1_g  = (const float*)d_in[6];  const float* ln1_b  = (const float*)d_in[7];
  const float* Wq = (const float*)d_in[8];  const float* bq = (const float*)d_in[9];
  const float* Wk = (const float*)d_in[10]; const float* bk = (const float*)d_in[11];
  const float* Wv = (const float*)d_in[12]; const float* bvp = (const float*)d_in[13];
  const float* Wo = (const float*)d_in[14]; const float* bo = (const float*)d_in[15];
  const float* ln2_g = (const float*)d_in[16]; const float* ln2_b = (const float*)d_in[17];
  const float* W1 = (const float*)d_in[18]; const float* b1 = (const float*)d_in[19];
  const float* W2 = (const float*)d_in[20]; const float* b2 = (const float*)d_in[21];
  const float* q_hop  = (const float*)d_in[22]; const float* q_edge = (const float*)d_in[23];
  const float* k_hop  = (const float*)d_in[24]; const float* k_edge = (const float*)d_in[25];
  const float* v_hop  = (const float*)d_in[26]; const float* v_edge = (const float*)d_in[27];
  const float* fln_g  = (const float*)d_in[28]; const float* fln_b  = (const float*)d_in[29];
  const float* out_W  = (const float*)d_in[30]; const float* out_b  = (const float*)d_in[31];

  // -------- workspace layout (~320 MB) --------
  char* ws = (char*)d_ws;
  size_t off = 0;
  auto alloc = [&](size_t bytes) { size_t r = off; off += (bytes + 255) & ~(size_t)255; return r; };
  auto US = [&](size_t o) { return (unsigned short*)(ws + o); };
  auto FP = [&](size_t o) { return (float*)(ws + o); };

  const size_t O_XB  = alloc((size_t)Bn*Nn*DIN*2);     // x bf16
  const size_t O_NW  = alloc((size_t)DIN*DM*2);        // node_W^T [256][128]
  const size_t O_WQ  = alloc((size_t)DM*DM*2);         // Wq^T
  const size_t O_WK  = alloc((size_t)DM*DM*2);
  const size_t O_WV  = alloc((size_t)DM*DM*2);
  const size_t O_WO  = alloc((size_t)DM*DM*2);
  const size_t O_W1  = alloc((size_t)DM*FFn*2);        // W1^T [1024][256]
  const size_t O_W2  = alloc((size_t)FFn*DM*2);        // W2^T [256][1024]
  const size_t O_OW  = alloc((size_t)DM*DOUT*2);       // out_W^T [128][256]
  const size_t O_QHT = alloc((size_t)NHOP*DM*2);       // q_hop bf16 (as-is)
  const size_t O_QET = alloc((size_t)NEDGE*DM*2);
  const size_t O_KHT = alloc((size_t)NHOP*DM*2);
  const size_t O_KET = alloc((size_t)NEDGE*DM*2);
  const size_t O_VHT = alloc((size_t)8*32*288*2);      // v_hop^T padded [8][32][288]
  const size_t O_VET = alloc((size_t)8*32*32*2);       // v_edge^T padded [8][32][32]
  const size_t O_H   = alloc((size_t)ROWS*DM*4);
  const size_t O_Y   = alloc((size_t)ROWS*DM*2);
  const size_t O_Q   = alloc((size_t)ROWS*DM*2);
  const size_t O_K   = alloc((size_t)ROWS*DM*2);
  const size_t O_V   = alloc((size_t)ROWS*DM*2);
  const size_t O_VT  = alloc((size_t)BH*32*512*2);     // v^T [bh][d][n]
  const size_t O_QH  = alloc((size_t)BH*Nn*NHOP*2);
  const size_t O_KH  = alloc((size_t)BH*Nn*NHOP*2);
  const size_t O_QE  = alloc((size_t)BH*Nn*NEDGE*2);
  const size_t O_KE  = alloc((size_t)BH*Nn*NEDGE*2);
  const size_t O_S   = alloc((size_t)BH*Nn*Nn*4);      // qk f32; att bf16 in-place
  const size_t O_VHA = alloc((size_t)BH*Nn*288*2);
  const size_t O_VEA = alloc((size_t)BH*Nn*32*2);
  const size_t O_CF  = alloc((size_t)ROWS*DM*4);       // ctx f32
  const size_t O_CB  = alloc((size_t)ROWS*DM*2);       // ctx bf16
  const size_t O_H2  = alloc((size_t)ROWS*DM*4);
  const size_t O_MID = alloc((size_t)ROWS*FFn*2);

  auto cvt = [&](const float* s, size_t o, int n) {
    cvt_bf16_k<<<dim3((n + 255) / 256), dim3(256), 0, stream>>>(s, US(o), n);
  };
  auto wtt = [&](const float* s, size_t o, int K, int N) {
    wt_t_k<<<dim3((K * N + 255) / 256), dim3(256), 0, stream>>>(s, US(o), K, N);
  };

  // 0) prep: convert + relayout all GEMM B operands (one-time, tiny)
  cvt(x, O_XB, Bn*Nn*DIN);
  wtt(node_W, O_NW, DIN, DM);
  wtt(Wq, O_WQ, DM, DM); wtt(Wk, O_WK, DM, DM); wtt(Wv, O_WV, DM, DM);
  wtt(Wo, O_WO, DM, DM);
  wtt(W1, O_W1, DM, FFn); wtt(W2, O_W2, FFn, DM); wtt(out_W, O_OW, DM, DOUT);
  cvt(q_hop, O_QHT, NHOP*DM); cvt(q_edge, O_QET, NEDGE*DM);
  cvt(k_hop, O_KHT, NHOP*DM); cvt(k_edge, O_KET, NEDGE*DM);
  vtab_t_k<<<dim3((8*32*288 + 255)/256), dim3(256), 0, stream>>>(v_hop, US(O_VHT), NHOP, 288);
  vtab_t_k<<<dim3((8*32*32  + 255)/256), dim3(256), 0, stream>>>(v_edge, US(O_VET), NEDGE, 32);

  // 1) h = x @ node_W + node_b        (32x64 wave tile, branch-free)
  run_gemm<2,4,false>(stream, US(O_XB),0,0,DIN, US(O_NW),0,0,DIN, node_b,
       nullptr,0,0,0, FP(O_H),0,0,DM, nullptr,0,0,0, ROWS,DM,DIN, 1,1,0);
  // 2) y = LN1(h)  (bf16)
  ln_k<<<dim3(ROWS/8), dim3(256), 0, stream>>>(FP(O_H), ln1_g, ln1_b, US(O_Y), ROWS);
  // 3) q,k,v = y @ W{q,k,v} + b  (bf16, layout [b,n,h,d] = [8192,256])
  run_gemm<2,4,false>(stream, US(O_Y),0,0,DM, US(O_WQ),0,0,DM, bq,
       nullptr,0,0,0, nullptr,0,0,0, US(O_Q),0,0,DM, ROWS,DM,DM, 1,1,0);
  run_gemm<2,4,false>(stream, US(O_Y),0,0,DM, US(O_WK),0,0,DM, bk,
       nullptr,0,0,0, nullptr,0,0,0, US(O_K),0,0,DM, ROWS,DM,DM, 1,1,0);
  run_gemm<2,4,false>(stream, US(O_Y),0,0,DM, US(O_WV),0,0,DM, bvp,
       nullptr,0,0,0, nullptr,0,0,0, US(O_V),0,0,DM, ROWS,DM,DM, 1,1,0);
  // 3b) v^T for the att@v GEMM
  v_t_k<<<dim3(BH*32*512/256), dim3(256), 0, stream>>>(US(O_V), US(O_VT));

  // 4) relative-position bias tables: qh/qe from q, kh/ke from k (K=32, 1 WMMA)
  const long sBb = (long)Nn*DM, sBh = 32;              // per-b / per-h act strides
  run_gemm<2,1,true>(stream, US(O_Q),sBb,sBh,DM, US(O_QHT),0,32,DM, nullptr,
       nullptr,0,0,0, nullptr,0,0,0, US(O_QH),(long)Hn*Nn*NHOP,(long)Nn*NHOP,NHOP,
       Nn,NHOP,32, Hn,BH,0);
  run_gemm<2,2,true>(stream, US(O_Q),sBb,sBh,DM, US(O_QET),0,32,DM, nullptr,
       nullptr,0,0,0, nullptr,0,0,0, US(O_QE),(long)Hn*Nn*NEDGE,(long)Nn*NEDGE,NEDGE,
       Nn,NEDGE,32, Hn,BH,0);
  run_gemm<2,1,true>(stream, US(O_K),sBb,sBh,DM, US(O_KHT),0,32,DM, nullptr,
       nullptr,0,0,0, nullptr,0,0,0, US(O_KH),(long)Hn*Nn*NHOP,(long)Nn*NHOP,NHOP,
       Nn,NHOP,32, Hn,BH,0);
  run_gemm<2,2,true>(stream, US(O_K),sBb,sBh,DM, US(O_KET),0,32,DM, nullptr,
       nullptr,0,0,0, nullptr,0,0,0, US(O_KE),(long)Hn*Nn*NEDGE,(long)Nn*NEDGE,NEDGE,
       Nn,NEDGE,32, Hn,BH,0);

  // 5) raw qk scores: S[bh,i,j] = q_i . k_j   (k is k-contiguous; N=512)
  run_gemm<2,4,false>(stream, US(O_Q),sBb,sBh,DM, US(O_K),sBb,sBh,DM, nullptr,
       nullptr,0,0,0, FP(O_S),(long)Hn*Nn*Nn,(long)Nn*Nn,Nn, nullptr,0,0,0,
       Nn,Nn,32, Hn,BH,0);

  // 6) bias gather + scale + mask + softmax -> att bf16 (in-place in O_S rows)
  softmax_bias_k<<<dim3(BH*Nn/8), dim3(256), 0, stream>>>(
      FP(O_S), US(O_S), US(O_QH), US(O_KH), US(O_QE), US(O_KE), dist, edge, mask);

  // 7) scatter att into hop/edge bins (zero-padded to 288/32)
  scatter_k<<<dim3(BH*Nn/8), dim3(256), 0, stream>>>(
      US(O_S), dist, edge, US(O_VHA), US(O_VEA));

  // 8) ctx = att@v + vha@Vh + vea@Ve   (f32 accumulate across 3 GEMMs; N=32)
  run_gemm<2,2,false>(stream, US(O_S),(long)Hn*Nn*1024,(long)Nn*1024,1024,
       US(O_VT),(long)Hn*32*512,(long)32*512,512, nullptr, nullptr,0,0,0,
       FP(O_CF),(long)Nn*DM,32,DM, nullptr,0,0,0, Nn,32,Nn, Hn,BH,0);
  run_gemm<2,2,false>(stream, US(O_VHA),(long)Hn*Nn*288,(long)Nn*288,288,
       US(O_VHT),0,(long)32*288,288, nullptr, FP(O_CF),(long)Nn*DM,32,DM,
       FP(O_CF),(long)Nn*DM,32,DM, nullptr,0,0,0, Nn,32,288, Hn,BH,0);
  run_gemm<2,2,false>(stream, US(O_VEA),(long)Hn*Nn*32,(long)Nn*32,32,
       US(O_VET),0,(long)32*32,32, nullptr, FP(O_CF),(long)Nn*DM,32,DM,
       FP(O_CF),(long)Nn*DM,32,DM, US(O_CB),(long)Nn*DM,32,DM,
       Nn,32,32, Hn,BH,0);

  // 9) h2 = h + ctx @ Wo + bo
  run_gemm<2,4,false>(stream, US(O_CB),0,0,DM, US(O_WO),0,0,DM, bo,
       FP(O_H),0,0,DM, FP(O_H2),0,0,DM, nullptr,0,0,0, ROWS,DM,DM, 1,1,0);
  // 10) y = LN2(h2)
  ln_k<<<dim3(ROWS/8), dim3(256), 0, stream>>>(FP(O_H2), ln2_g, ln2_b, US(O_Y), ROWS);
  // 11) mid = gelu(y @ W1 + b1)  (bf16)
  run_gemm<2,4,false>(stream, US(O_Y),0,0,DM, US(O_W1),0,0,DM, b1,
       nullptr,0,0,0, nullptr,0,0,0, US(O_MID),0,0,FFn, ROWS,FFn,DM, 1,1,1);
  // 12) h3 = h2 + mid @ W2 + b2   (reuse O_H)
  run_gemm<2,4,false>(stream, US(O_MID),0,0,FFn, US(O_W2),0,0,FFn, b2,
       FP(O_H2),0,0,DM, FP(O_H),0,0,DM, nullptr,0,0,0, ROWS,DM,FFn, 1,1,0);
  // 13) y = LN_final(h3)
  ln_k<<<dim3(ROWS/8), dim3(256), 0, stream>>>(FP(O_H), fln_g, fln_b, US(O_Y), ROWS);
  // 14) out = y @ out_W + out_b  -> d_out (f32; N=128)
  run_gemm<2,4,false>(stream, US(O_Y),0,0,DM, US(O_OW),0,0,DM, out_b,
       nullptr,0,0,0, (float*)d_out,0,0,DOUT, nullptr,0,0,0, ROWS,DOUT,DM, 1,1,0);
}
